// Decoder_14645838479787
// MI455X (gfx1250) — compile-verified, matrix operand order checked by reading, not run on hardware
//
#include <hip/hip_runtime.h>

typedef _Float16 h16;
typedef __attribute__((ext_vector_type(16))) _Float16 v16h;
typedef __attribute__((ext_vector_type(8)))  float    v8f;

#define BB   4096
#define HH   64
#define GG   256
#define TT   64
#define OUTN 310
#define NPAD 320

// ---- workspace byte offsets ----
#define W0X_OFF   0u          // 256*64 f16 = 32768
#define WHH0_OFF  32768u
#define WIH1_OFF  65536u
#define WHH1_OFF  98304u
#define FCW_OFF   131072u     // 320*64 f16 = 40960
#define B0_OFF    172032u     // 256 f32
#define B1_OFF    173056u     // 256 f32
#define FCB_OFF   174080u     // 320 f32 (padded to 1280B)
#define V_OFF     175360u     // 64*4096*64 f16 = 33554432

// ---- dynamic LDS byte offsets (recurrence kernel) ----
#define L_W      0u           // 4 * 256*64 f16 = 131072
#define L_GATES  131072u      // 32*256 f32     = 32768
#define L_VBUF   163840u      // 32*64 f16      = 4096
#define L_H0BUF  167936u
#define L_H1BUF  172032u
#define L_BIAS   176128u      // 512 f32        = 2048
#define SMEM_BYTES 178176u

__device__ __forceinline__ float sigm(float x) { return 1.0f / (1.0f + __expf(-x)); }

union FragU { uint4 u[2]; v16h v; };
union Pack8 { h16 h[8]; uint4 u; };

// Load one 16x16x32 A/B fragment. Layout (ISA 7.12.2, 16-bit):
// lanes 0-15:  halves[0..7]=K+0..7,  halves[8..15]=K+16..23
// lanes 16-31: halves[0..7]=K+8..15, halves[8..15]=K+24..31
// Caller folds the lane-dependent +0/+8 (hiK) into p.
__device__ __forceinline__ v16h ldfrag(const h16* p) {
  FragU f;
  f.u[0] = *(const uint4*)p;
  f.u[1] = *(const uint4*)(p + 16);
  return f.v;
}

__device__ __forceinline__ v8f wmma_f16(v16h a, v16h b, v8f c) {
  return __builtin_amdgcn_wmma_f32_16x16x32_f16(false, a, false, b, (short)0, c,
                                                false, false);
}

// One LSTM gate GEMM: gates[32x256] = A0[32x64] @ W0^T + A1[32x64] @ W1^T
// wave wv handles m-tile (wv&1), n-tiles 4*(wv>>1)+0..3
__device__ __forceinline__ void layer_gemm(const h16* __restrict__ bufA0,
                                           const h16* __restrict__ W0,
                                           const h16* __restrict__ bufA1,
                                           const h16* __restrict__ W1,
                                           float* __restrict__ gates,
                                           int mt, int ng, int laneLow, int hiK) {
  const h16* a0p = bufA0 + (mt * 16 + laneLow) * 64 + hiK;
  const h16* a1p = bufA1 + (mt * 16 + laneLow) * 64 + hiK;
  v16h a00 = ldfrag(a0p), a01 = ldfrag(a0p + 32);
  v16h a10 = ldfrag(a1p), a11 = ldfrag(a1p + 32);
#pragma unroll
  for (int i = 0; i < 4; ++i) {
    const int nt = ng * 4 + i;
    const h16* b0p = W0 + (nt * 16 + laneLow) * 64 + hiK;
    const h16* b1p = W1 + (nt * 16 + laneLow) * 64 + hiK;
    v16h b00 = ldfrag(b0p), b01 = ldfrag(b0p + 32);
    v16h b10 = ldfrag(b1p), b11 = ldfrag(b1p + 32);
    v8f c = {0.f, 0.f, 0.f, 0.f, 0.f, 0.f, 0.f, 0.f};
    c = wmma_f16(a00, b00, c);
    c = wmma_f16(a01, b01, c);
    c = wmma_f16(a10, b10, c);
    c = wmma_f16(a11, b11, c);
    // D layout: lanes0-15 vgpr r -> M=r; lanes16-31 -> M=r+8; N = lane&15
    float* g = gates + (mt * 16 + hiK) * GG + nt * 16 + laneLow;
#pragma unroll
    for (int r = 0; r < 8; ++r) g[r * GG] = c[r];
  }
}

// ---------------- prep kernels ----------------
__global__ void prep_w0x(const float* __restrict__ w_ih0, const float* __restrict__ fc_w,
                         const float* __restrict__ b_ih0, const float* __restrict__ b_hh0,
                         const float* __restrict__ fc_b, h16* __restrict__ W0X,
                         float* __restrict__ b0) {
  const int r = blockIdx.x;     // 0..255
  const int c = threadIdx.x;    // 0..63
  const float* wr = w_ih0 + r * OUTN;
  float acc = 0.f;
  for (int k = 0; k < OUTN; ++k) acc += wr[k] * fc_w[k * HH + c];
  W0X[r * HH + c] = (h16)acc;
  if (c == 0) {
    float bb = b_ih0[r] + b_hh0[r];
    for (int k = 0; k < OUTN; ++k) bb += wr[k] * fc_b[k];
    b0[r] = bb;
  }
}

__global__ void prep_misc(const float* __restrict__ w_hh0, const float* __restrict__ w_ih1,
                          const float* __restrict__ w_hh1, const float* __restrict__ fc_w,
                          const float* __restrict__ b_ih1, const float* __restrict__ b_hh1,
                          const float* __restrict__ fc_b, const float* __restrict__ input,
                          h16* __restrict__ WHH0, h16* __restrict__ WIH1,
                          h16* __restrict__ WHH1, h16* __restrict__ FCW,
                          float* __restrict__ B1, float* __restrict__ FCB,
                          h16* __restrict__ V0) {
  int idx = blockIdx.x * 256 + threadIdx.x;
  if (idx < 16384) { WHH0[idx] = (h16)w_hh0[idx]; return; }
  idx -= 16384;
  if (idx < 16384) { WIH1[idx] = (h16)w_ih1[idx]; return; }
  idx -= 16384;
  if (idx < 16384) { WHH1[idx] = (h16)w_hh1[idx]; return; }
  idx -= 16384;
  if (idx < NPAD * HH) {
    int n = idx >> 6, k = idx & 63;
    FCW[idx] = (h16)((n < OUTN) ? fc_w[n * HH + k] : 0.f);
    return;
  }
  idx -= NPAD * HH;
  if (idx < GG) { B1[idx] = b_ih1[idx] + b_hh1[idx]; return; }
  idx -= GG;
  if (idx < NPAD) { FCB[idx] = (idx < OUTN) ? fc_b[idx] : 0.f; return; }
  idx -= NPAD;
  if (idx < BB * HH) { V0[idx] = (h16)input[idx]; return; }
}

// ---------------- recurrence kernel ----------------
__global__ __launch_bounds__(256)
void lstm_rec(const float* __restrict__ input, const float* __restrict__ hidden,
              const float* __restrict__ cell, const h16* __restrict__ wsrc,
              const float* __restrict__ b0g, const float* __restrict__ b1g,
              h16* __restrict__ V, float* __restrict__ out) {
  extern __shared__ char smem[];
  h16*   wlds  = (h16*)(smem + L_W);
  float* gates = (float*)(smem + L_GATES);
  h16*   vbuf  = (h16*)(smem + L_VBUF);
  h16*   h0buf = (h16*)(smem + L_H0BUF);
  h16*   h1buf = (h16*)(smem + L_H1BUF);
  float* bias  = (float*)(smem + L_BIAS);

  const int tid = threadIdx.x;
  const int lane = tid & 31;
  const int wv = tid >> 5;
  const int laneLow = lane & 15;
  const int hiK = (lane >> 4) << 3;   // 0 or 8
  const int mt = wv & 1;
  const int ng = wv >> 1;

  // Stage weights (4 contiguous 256x64 f16 matrices) into LDS
  {
    const uint4* src = (const uint4*)wsrc;
    uint4* dst = (uint4*)wlds;
    for (int i = tid; i < 8192; i += 256) dst[i] = src[i];
    bias[tid] = b0g[tid];
    bias[256 + tid] = b1g[tid];
  }

  const int crow = tid >> 3;              // 0..31
  const int ccol0 = (tid & 7) * 8;        // 0,8,..,56
  const int rowg = blockIdx.x * 32 + crow;

  float c0reg[8], c1reg[8];
  {
    Pack8 pv, p0, p1;
#pragma unroll
    for (int j = 0; j < 8; ++j) {
      const int col = ccol0 + j;
      pv.h[j] = (h16)input[rowg * HH + col];
      p0.h[j] = (h16)hidden[rowg * HH + col];
      p1.h[j] = (h16)hidden[BB * HH + rowg * HH + col];
      c0reg[j] = cell[rowg * HH + col];
      c1reg[j] = cell[BB * HH + rowg * HH + col];
    }
    *(uint4*)(vbuf + crow * 64 + ccol0) = pv.u;
    *(uint4*)(h0buf + crow * 64 + ccol0) = p0.u;
    *(uint4*)(h1buf + crow * 64 + ccol0) = p1.u;
  }
  __syncthreads();

  const h16* W0X  = wlds;
  const h16* WHH0 = wlds + 16384;
  const h16* WIH1 = wlds + 32768;
  const h16* WHH1 = wlds + 49152;

  const size_t hOff = (size_t)BB * TT * OUTN;
  const size_t cOff = hOff + 2ull * BB * HH;

  for (int t = 1; t < TT; ++t) {
    const bool last = (t == TT - 1);
    // layer 0: g0 = v_prev @ W0X^T + h0 @ WHH0^T
    layer_gemm(vbuf, W0X, h0buf, WHH0, gates, mt, ng, laneLow, hiK);
    __syncthreads();
    {
      Pack8 ph;
      const float* g = gates + crow * GG;
#pragma unroll
      for (int j = 0; j < 8; ++j) {
        const int col = ccol0 + j;
        const float gi = g[col] + bias[col];
        const float gf = g[64 + col] + bias[64 + col];
        const float gz = g[128 + col] + bias[128 + col];
        const float go = g[192 + col] + bias[192 + col];
        const float cn = sigm(gf) * c0reg[j] + sigm(gi) * tanhf(gz);
        const float hn = sigm(go) * tanhf(cn);
        c0reg[j] = cn;
        ph.h[j] = (h16)hn;
        if (last) {
          out[hOff + (size_t)rowg * HH + col] = hn;
          out[cOff + (size_t)rowg * HH + col] = cn;
        }
      }
      *(uint4*)(h0buf + crow * 64 + ccol0) = ph.u;
    }
    __syncthreads();
    // layer 1: g1 = h0_new @ WIH1^T + h1_old @ WHH1^T
    layer_gemm(h0buf, WIH1, h1buf, WHH1, gates, mt, ng, laneLow, hiK);
    __syncthreads();
    {
      Pack8 ph;
      const float* g = gates + crow * GG;
#pragma unroll
      for (int j = 0; j < 8; ++j) {
        const int col = ccol0 + j;
        const float gi = g[col] + bias[256 + col];
        const float gf = g[64 + col] + bias[256 + 64 + col];
        const float gz = g[128 + col] + bias[256 + 128 + col];
        const float go = g[192 + col] + bias[256 + 192 + col];
        const float cn = sigm(gf) * c1reg[j] + sigm(gi) * tanhf(gz);
        const float hn = sigm(go) * tanhf(cn);
        c1reg[j] = cn;
        ph.h[j] = (h16)hn;
        if (last) {
          out[hOff + (size_t)BB * HH + (size_t)rowg * HH + col] = hn;
          out[cOff + (size_t)BB * HH + (size_t)rowg * HH + col] = cn;
        }
      }
      *(uint4*)(h1buf + crow * 64 + ccol0) = ph.u;
      *(uint4*)(vbuf + crow * 64 + ccol0) = ph.u;
      *(uint4*)(V + ((size_t)t * BB + rowg) * HH + ccol0) = ph.u;
    }
    __syncthreads();
  }
}

// ---------------- output GEMM: outs = V @ fcw^T + fcb, time-reversed ----------------
__global__ __launch_bounds__(256)
void fc_out(const h16* __restrict__ V, const h16* __restrict__ fcw,
            const float* __restrict__ fcb, float* __restrict__ out) {
  const int tid = threadIdx.x;
  const int lane = tid & 31;
  const int wv = tid >> 5;
  const int laneLow = lane & 15;
  const int hiK = (lane >> 4) << 3;
  const int wg = blockIdx.x * 8 + wv;       // < 16384*20
  const int mtile = wg / 20;
  const int ntile = wg - mtile * 20;

  const h16* ap = V + ((size_t)(mtile * 16 + laneLow)) * HH + hiK;
  v16h a0 = ldfrag(ap), a1 = ldfrag(ap + 32);
  const h16* bp = fcw + (ntile * 16 + laneLow) * HH + hiK;
  v16h b0 = ldfrag(bp), b1 = ldfrag(bp + 32);
  v8f c = {0.f, 0.f, 0.f, 0.f, 0.f, 0.f, 0.f, 0.f};
  c = wmma_f16(a0, b0, c);
  c = wmma_f16(a1, b1, c);

  const int n = ntile * 16 + laneLow;       // output column (<320, write if <310)
  const float bv = fcb[n];
  const int rbase = mtile * 16 + hiK;
#pragma unroll
  for (int r = 0; r < 8; ++r) {
    const int rowM = rbase + r;             // = s*4096 + b
    const int s = rowM >> 12;
    const int b = rowM & 4095;
    if (n < OUTN)
      out[(size_t)b * TT * OUTN + (size_t)(63 - s) * OUTN + n] = c[r] + bv;
  }
}

extern "C" void kernel_launch(void* const* d_in, const int* in_sizes, int n_in,
                              void* d_out, int out_size, void* d_ws, size_t ws_size,
                              hipStream_t stream) {
  (void)in_sizes; (void)n_in; (void)out_size; (void)ws_size;
  const float* input  = (const float*)d_in[0];
  const float* hidden = (const float*)d_in[1];
  const float* cell   = (const float*)d_in[2];
  const float* w_ih0  = (const float*)d_in[4];
  const float* w_hh0  = (const float*)d_in[5];
  const float* b_ih0  = (const float*)d_in[6];
  const float* b_hh0  = (const float*)d_in[7];
  const float* w_ih1  = (const float*)d_in[8];
  const float* w_hh1  = (const float*)d_in[9];
  const float* b_ih1  = (const float*)d_in[10];
  const float* b_hh1  = (const float*)d_in[11];
  const float* fc_w   = (const float*)d_in[12];
  const float* fc_b   = (const float*)d_in[13];
  float* out = (float*)d_out;
  char* ws = (char*)d_ws;

  h16*   W0X  = (h16*)(ws + W0X_OFF);
  h16*   WHH0 = (h16*)(ws + WHH0_OFF);
  h16*   WIH1 = (h16*)(ws + WIH1_OFF);
  h16*   WHH1 = (h16*)(ws + WHH1_OFF);
  h16*   FCW  = (h16*)(ws + FCW_OFF);
  float* B0   = (float*)(ws + B0_OFF);
  float* B1   = (float*)(ws + B1_OFF);
  float* FCB  = (float*)(ws + FCB_OFF);
  h16*   V    = (h16*)(ws + V_OFF);

  prep_w0x<<<256, 64, 0, stream>>>(w_ih0, fc_w, b_ih0, b_hh0, fc_b, W0X, B0);
  prep_misc<<<1299, 256, 0, stream>>>(w_hh0, w_ih1, w_hh1, fc_w, b_ih1, b_hh1,
                                      fc_b, input, WHH0, WIH1, WHH1, FCW, B1, FCB, V);
  lstm_rec<<<128, 256, SMEM_BYTES, stream>>>(input, hidden, cell, (const h16*)ws,
                                             B0, B1, V, out);
  fc_out<<<40960, 256, 0, stream>>>(V, FCW, FCB, out);
}
